// DeepSeekMoE_22239340659168
// MI455X (gfx1250) — matrix operand order repaired and checked by reading, not compiled, not run on hardware
//
#include <hip/hip_runtime.h>
#include <cstdint>
#include <cstddef>

typedef __attribute__((ext_vector_type(16))) __bf16 v16bf;
typedef __attribute__((ext_vector_type(8)))  __bf16 v8bf;
typedef __attribute__((ext_vector_type(8)))  float  v8f;

#define D_MODEL 1024
#define D_HID   1408
#define NTOK    2048
#define EPS_F   1.1920928955078125e-7f

#define BM 128
#define BN 128
#define BK 32
#define BKP 40   // padded LDS row (80B): 16B-aligned, conflict-free b128 frag reads

// ------------------------------ RMSNorm -> bf16 ------------------------------
__global__ __launch_bounds__(256) void k_rmsnorm(const float* __restrict__ x,
                                                 const float* __restrict__ nw,
                                                 __bf16* __restrict__ xn) {
  int tok = blockIdx.x;
  const float* xr = x + (size_t)tok * D_MODEL;
  float v[4]; float sq = 0.f;
  #pragma unroll
  for (int j = 0; j < 4; ++j) { v[j] = xr[threadIdx.x + 256 * j]; sq += v[j] * v[j]; }
  #pragma unroll
  for (int off = 16; off > 0; off >>= 1) sq += __shfl_xor(sq, off, 32);
  __shared__ float wsum[8];
  int wave = threadIdx.x >> 5, lane = threadIdx.x & 31;
  if (lane == 0) wsum[wave] = sq;
  __syncthreads();
  float tot = 0.f;
  #pragma unroll
  for (int w = 0; w < 8; ++w) tot += wsum[w];
  float scale = rsqrtf(tot * (1.0f / D_MODEL) + EPS_F);
  #pragma unroll
  for (int j = 0; j < 4; ++j) {
    int d = threadIdx.x + 256 * j;
    xn[(size_t)tok * D_MODEL + d] = (__bf16)(v[j] * scale * nw[d]);
  }
}

// ---------------- transpose-convert fp32 [R][C] -> bf16 [C-major][R] ----------
// out[z*z_off + c*out_ld + r] = in[z*R*C + r*C + c]
__global__ void k_pack(const float* __restrict__ in, __bf16* __restrict__ out,
                       int R, int C, int out_ld, long z_off) {
  __shared__ float tile[32][33];
  int z = blockIdx.z;
  const float* inz = in + (size_t)z * R * C;
  int c0 = blockIdx.x * 32, r0 = blockIdx.y * 32;
  #pragma unroll
  for (int k = 0; k < 4; ++k)
    tile[threadIdx.y + 8 * k][threadIdx.x] =
        inz[(size_t)(r0 + threadIdx.y + 8 * k) * C + c0 + threadIdx.x];
  __syncthreads();
  #pragma unroll
  for (int k = 0; k < 4; ++k)
    out[(size_t)z * z_off + (size_t)(c0 + threadIdx.y + 8 * k) * out_ld + (r0 + threadIdx.x)] =
        (__bf16)tile[threadIdx.x][threadIdx.y + 8 * k];
}

// ---------------------------- summed shared bias -----------------------------
__global__ void k_bsum(const float* __restrict__ bs2, float* __restrict__ bsum) {
  int d = blockIdx.x * 256 + threadIdx.x;
  bsum[d] = bs2[d] + bs2[D_MODEL + d];
}

// --------------------------------- Router ------------------------------------
__global__ __launch_bounds__(32) void k_router(const float* __restrict__ x,
                                               const float* __restrict__ nw,
                                               const float* __restrict__ Wr,
                                               int* __restrict__ cnt, int* __restrict__ list,
                                               float* __restrict__ gate, int* __restrict__ slot) {
  int tok = blockIdx.x;
  int lane = threadIdx.x;
  const float* xr = x + (size_t)tok * D_MODEL;
  float sq = 0.f;
  float dot[8] = {0.f, 0.f, 0.f, 0.f, 0.f, 0.f, 0.f, 0.f};
  for (int j = 0; j < D_MODEL / 32; ++j) {
    int d = lane + j * 32;
    float v = xr[d];
    sq += v * v;
    float vw = v * nw[d];
    #pragma unroll
    for (int e = 0; e < 8; ++e) dot[e] += vw * Wr[e * D_MODEL + d];
  }
  #pragma unroll
  for (int off = 16; off > 0; off >>= 1) {
    sq += __shfl_xor(sq, off, 32);
    #pragma unroll
    for (int e = 0; e < 8; ++e) dot[e] += __shfl_xor(dot[e], off, 32);
  }
  if (lane == 0) {
    float scale = rsqrtf(sq * (1.0f / D_MODEL) + EPS_F);
    float l[8];
    #pragma unroll
    for (int e = 0; e < 8; ++e) l[e] = dot[e] * scale;
    int i1 = 0; float b1v = l[0];
    #pragma unroll
    for (int e = 1; e < 8; ++e) if (l[e] > b1v) { b1v = l[e]; i1 = e; }  // jax tie-break: lowest idx
    int i2 = -1; float b2v = -3.4e38f;
    #pragma unroll
    for (int e = 0; e < 8; ++e) if (e != i1 && l[e] > b2v) { b2v = l[e]; i2 = e; }
    int p = atomicAdd(&cnt[i1], 1);
    list[i1 * NTOK + p] = tok; gate[i1 * NTOK + p] = b1v; slot[i1 * NTOK + p] = 0;
    p = atomicAdd(&cnt[i2], 1);
    list[i2 * NTOK + p] = tok; gate[i2 * NTOK + p] = b2v; slot[i2 * NTOK + p] = 1;
  }
}

// ------------------------------ WMMA GEMM core --------------------------------
// C[M,N] = A[M,K](bf16, row-major, ld=lda) x Bt[N,K](bf16, "B^T", ld=ldb) + bias
// MODE 0: silu -> bf16 out (shared GEMM1)       MODE 1: +bias -> f32 out (shared GEMM2)
// MODE 2: gather-A rows, silu -> bf16 (routed1) MODE 3: scatter gate*(acc+bias) (routed2)
__device__ __forceinline__ float silu_f(float v) { return v / (1.0f + __expf(-v)); }

__device__ __forceinline__ unsigned lds_addr32(const void* p) {
  // generic pointer to LDS: addr[31:0] is the LDS byte offset (aperture in high bits)
  return (unsigned)(size_t)p;
}

__device__ __forceinline__ void async_ld_b128(unsigned lds_off, const void* gp) {
  asm volatile("global_load_async_to_lds_b128 %0, %1, off"
               :: "v"(lds_off), "v"((unsigned long long)(size_t)gp)
               : "memory");
}

__device__ __forceinline__ void wait_asynccnt0() {
  asm volatile("s_wait_asynccnt 0x0" ::: "memory");
}

template <int MODE>
__global__ __launch_bounds__(256) void k_gemm(
    const __bf16* __restrict__ A, int lda, long A_zoff,
    const __bf16* __restrict__ Bt, int ldb, long B_zoff,
    const float* __restrict__ bias, int bias_zoff,
    float* __restrict__ outF, __bf16* __restrict__ outH, int ldo, long out_zoff,
    int M, int N, int K,
    const int* __restrict__ cnts, const int* __restrict__ lists,
    const float* __restrict__ gates, const int* __restrict__ slots) {
  int e = blockIdx.z;
  const __bf16* Ae = A + (size_t)e * A_zoff;
  const __bf16* Be = Bt + (size_t)e * B_zoff;
  const float* biasE = bias + (size_t)e * bias_zoff;
  int Mv = M;
  const int* list = nullptr;
  if (MODE == 2 || MODE == 3) { Mv = cnts[e]; list = lists + e * NTOK; }
  int m0 = blockIdx.y * BM;
  int n0 = blockIdx.x * BN;
  if ((MODE == 2 || MODE == 3) && m0 >= Mv) return;  // uniform exit

  __shared__ __bf16 As[2][BM * BKP];
  __shared__ __bf16 Bs[2][BN * BKP];

  int wave = threadIdx.x >> 5;
  int lane = threadIdx.x & 31;
  int r = lane & 15, hi = lane >> 4;
  int wm = (wave >> 2) * 64;  // 2 wave-rows (64 M each)
  int wn = (wave & 3) * 32;   // 4 wave-cols (32 N each)

  // async global->LDS tile stage (no VGPR staging, tracked by ASYNCcnt)
  auto load_tiles = [&](int buf, int k0) {
    #pragma unroll
    for (int i = 0; i < 2; ++i) {
      int idx = threadIdx.x + i * 256;
      int row = idx >> 2;
      int col = (idx & 3) * 8;
      int gm = m0 + row;
      unsigned la = lds_addr32(&As[buf][row * BKP + col]);
      if (MODE == 2 || MODE == 3) {
        if (gm < Mv) {
          size_t arow = (MODE == 2) ? (size_t)list[gm] : (size_t)gm;
          async_ld_b128(la, Ae + arow * lda + k0 + col);
        } else {
          uint4 z = {0u, 0u, 0u, 0u};
          *(uint4*)(&As[buf][row * BKP + col]) = z;  // DS store; covered by barrier's DS wait
        }
      } else {
        async_ld_b128(la, Ae + (size_t)gm * lda + k0 + col);
      }
      unsigned lb = lds_addr32(&Bs[buf][row * BKP + col]);
      async_ld_b128(lb, Be + (size_t)(n0 + row) * ldb + k0 + col);
    }
  };

  v8f acc[4][2] = {};

  load_tiles(0, 0);
  wait_asynccnt0();
  __syncthreads();

  int cur = 0;
  for (int k0 = 0; k0 < K; k0 += BK) {
    if (k0 + BK < K) load_tiles(cur ^ 1, k0 + BK);  // stream next tile during compute

    v16bf afr[4], bfr[2];
    #pragma unroll
    for (int mi = 0; mi < 4; ++mi) {  // A 16x32: lanes<16: K 0-7 & 16-23; lanes>=16: K 8-15 & 24-31
      int row = wm + mi * 16 + r;
      union { v16bf v; v8bf h[2]; } u;
      u.h[0] = *(const v8bf*)(&As[cur][row * BKP + hi * 8]);
      u.h[1] = *(const v8bf*)(&As[cur][row * BKP + hi * 8 + 16]);
      afr[mi] = u.v;
    }
    #pragma unroll
    for (int ni = 0; ni < 2; ++ni) {  // B 32x16 (as N-major): lanes<16: K 0-7 & 16-23; lanes>=16: K 8-15 & 24-31
      int row = wn + ni * 16 + r;
      union { v16bf v; v8bf h[2]; } u;
      u.h[0] = *(const v8bf*)(&Bs[cur][row * BKP + hi * 8]);
      u.h[1] = *(const v8bf*)(&Bs[cur][row * BKP + hi * 8 + 16]);
      bfr[ni] = u.v;
    }
    #pragma unroll
    for (int mi = 0; mi < 4; ++mi)
      #pragma unroll
      for (int ni = 0; ni < 2; ++ni)
        acc[mi][ni] = __builtin_amdgcn_wmma_f32_16x16x32_bf16(
            false, afr[mi], false, bfr[ni], (short)0, acc[mi][ni], false, false);

    wait_asynccnt0();   // next buffer resident (this wave's issues)
    __syncthreads();    // all waves done reading cur + done filling nxt
    cur ^= 1;
  }

  // Epilogue: D layout — VGPR i holds (m = 8*hi + i, n = r) within each 16x16 frag
  #pragma unroll
  for (int mi = 0; mi < 4; ++mi) {
    #pragma unroll
    for (int ni = 0; ni < 2; ++ni) {
      #pragma unroll
      for (int i = 0; i < 8; ++i) {
        int gr = m0 + wm + mi * 16 + 8 * hi + i;
        int gc = n0 + wn + ni * 16 + r;
        float val = acc[mi][ni][i] + biasE[gc];
        if (MODE == 0) {
          outH[(size_t)gr * ldo + gc] = (__bf16)silu_f(val);
        } else if (MODE == 1) {
          outF[(size_t)gr * ldo + gc] = val;
        } else if (MODE == 2) {
          if (gr < Mv) outH[(size_t)e * out_zoff + (size_t)gr * ldo + gc] = (__bf16)silu_f(val);
        } else {
          if (gr < Mv) {
            int tk = list[gr];
            float g = gates[e * NTOK + gr];
            int s = slots[e * NTOK + gr];
            outF[((size_t)tk * 2 + s) * D_MODEL + gc] = g * val;
          }
        }
      }
    }
  }
}

// -------------------------------- Combine -------------------------------------
__global__ __launch_bounds__(256) void k_combine(const float* __restrict__ x,
                                                 const float* __restrict__ sh,
                                                 const float* __restrict__ rout,
                                                 float* __restrict__ out) {
  size_t i = (size_t)blockIdx.x * 256 + threadIdx.x;
  size_t tok = i >> 10;
  size_t d = i & 1023;
  out[i] = x[i] + sh[i] + rout[(tok * 2) * D_MODEL + d] + rout[(tok * 2 + 1) * D_MODEL + d];
}

// ------------------------------- Launcher --------------------------------------
extern "C" void kernel_launch(void* const* d_in, const int* in_sizes, int n_in,
                              void* d_out, int out_size, void* d_ws, size_t ws_size,
                              hipStream_t stream) {
  const float* x    = (const float*)d_in[0];
  const float* nw   = (const float*)d_in[1];
  const float* Wr   = (const float*)d_in[2];
  const float* Ws1  = (const float*)d_in[3];
  const float* bs1  = (const float*)d_in[4];
  const float* Ws2  = (const float*)d_in[5];
  const float* bs2  = (const float*)d_in[6];
  const float* W1   = (const float*)d_in[7];
  const float* b1   = (const float*)d_in[8];
  const float* W2   = (const float*)d_in[9];
  const float* b2   = (const float*)d_in[10];
  float* out = (float*)d_out;

  char* base = (char*)d_ws;
  size_t o = 0;
  auto take = [&](size_t bytes) { size_t r = o; o = (o + bytes + 255) & ~(size_t)255; return r; };

  __bf16* xn    = (__bf16*)(base + take((size_t)NTOK * D_MODEL * 2));
  __bf16* Ws1T  = (__bf16*)(base + take((size_t)2 * D_HID * D_MODEL * 2));   // [2816][1024]
  __bf16* Ws2T  = (__bf16*)(base + take((size_t)D_MODEL * 2 * D_HID * 2));   // [1024][2816]
  __bf16* W1T   = (__bf16*)(base + take((size_t)8 * D_HID * D_MODEL * 2));   // [8][1408][1024]
  __bf16* W2T   = (__bf16*)(base + take((size_t)8 * D_MODEL * D_HID * 2));   // [8][1024][1408]
  __bf16* Hs    = (__bf16*)(base + take((size_t)NTOK * 2 * D_HID * 2));      // [2048][2816]
  __bf16* Hr    = (__bf16*)(base + take((size_t)8 * NTOK * D_HID * 2));      // [8][2048][1408]
  float*  shout = (float*) (base + take((size_t)NTOK * D_MODEL * 4));
  float*  rout  = (float*) (base + take((size_t)NTOK * 2 * D_MODEL * 4));
  float*  bsum  = (float*) (base + take((size_t)D_MODEL * 4));
  int*    cnt   = (int*)   (base + take(8 * 4));
  int*    list  = (int*)   (base + take((size_t)8 * NTOK * 4));
  float*  gate  = (float*) (base + take((size_t)8 * NTOK * 4));
  int*    slot  = (int*)   (base + take((size_t)8 * NTOK * 4));

  hipMemsetAsync(cnt, 0, 8 * sizeof(int), stream);

  k_rmsnorm<<<NTOK, 256, 0, stream>>>(x, nw, xn);

  dim3 pb(32, 8);
  k_pack<<<dim3(D_HID / 32, D_MODEL / 32, 2), pb, 0, stream>>>(Ws1, Ws1T, D_MODEL, D_HID, D_MODEL, (long)D_HID * D_MODEL);
  k_pack<<<dim3(D_MODEL / 32, D_HID / 32, 2), pb, 0, stream>>>(Ws2, Ws2T, D_HID, D_MODEL, 2 * D_HID, (long)D_HID);
  k_pack<<<dim3(D_HID / 32, D_MODEL / 32, 8), pb, 0, stream>>>(W1, W1T, D_MODEL, D_HID, D_MODEL, (long)D_HID * D_MODEL);
  k_pack<<<dim3(D_MODEL / 32, D_HID / 32, 8), pb, 0, stream>>>(W2, W2T, D_HID, D_MODEL, D_HID, (long)D_MODEL * D_HID);

  k_bsum<<<D_MODEL / 256, 256, 0, stream>>>(bs2, bsum);
  k_router<<<NTOK, 32, 0, stream>>>(x, nw, Wr, cnt, list, gate, slot);

  // Shared GEMM1: [2048,1024] x [1024,2816] -> silu -> Hs
  k_gemm<0><<<dim3(2 * D_HID / BN, NTOK / BM, 1), 256, 0, stream>>>(
      xn, D_MODEL, 0L, Ws1T, D_MODEL, 0L, bs1, 0, nullptr, Hs, 2 * D_HID, 0L,
      NTOK, 2 * D_HID, D_MODEL, nullptr, nullptr, nullptr, nullptr);
  // Shared GEMM2: [2048,2816] x [2816,1024] -> shout
  k_gemm<1><<<dim3(D_MODEL / BN, NTOK / BM, 1), 256, 0, stream>>>(
      Hs, 2 * D_HID, 0L, Ws2T, 2 * D_HID, 0L, bsum, 0, shout, nullptr, D_MODEL, 0L,
      NTOK, D_MODEL, 2 * D_HID, nullptr, nullptr, nullptr, nullptr);
  // Routed GEMM1 (gathered): xn[list] x W1T[e] -> silu -> Hr[e]
  k_gemm<2><<<dim3(D_HID / BN, NTOK / BM, 8), 256, 0, stream>>>(
      xn, D_MODEL, 0L, W1T, D_MODEL, (long)D_HID * D_MODEL, b1, D_HID,
      nullptr, Hr, D_HID, (long)NTOK * D_HID,
      NTOK, D_HID, D_MODEL, cnt, list, nullptr, nullptr);
  // Routed GEMM2: Hr[e] x W2T[e] -> gate-scaled scatter into rout[token][slot]
  k_gemm<3><<<dim3(D_MODEL / BN, NTOK / BM, 8), 256, 0, stream>>>(
      Hr, D_HID, (long)NTOK * D_HID, W2T, D_HID, (long)D_MODEL * D_HID, b2, D_MODEL,
      rout, nullptr, D_MODEL, 0L,
      NTOK, D_MODEL, D_HID, cnt, list, gate, slot);

  k_combine<<<(NTOK * D_MODEL) / 256, 256, 0, stream>>>(x, shout, rout, out);
}
